// ContentBasedAttention_27178553049488
// MI455X (gfx1250) — matrix operand order repaired and checked by reading, not compiled, not run on hardware
//
#include <hip/hip_runtime.h>
#include <math.h>

typedef __attribute__((ext_vector_type(16))) __bf16 v16bf;
typedef __attribute__((ext_vector_type(8)))  float  v8f;

#define B_ 32
#define T_ 2048
#define H_ 1024
#define C_ 1024
#define LDS_STRIDE 1032   // 1024 + 8 ushorts (16B) pad -> conflict-free ds_load_b128

__device__ __forceinline__ unsigned short f2bf(float f) {
  unsigned u = __float_as_uint(f);
  u += 0x7FFFu + ((u >> 16) & 1u);        // round-to-nearest-even
  return (unsigned short)(u >> 16);
}

// V [H,C] f32 row-major  ->  Vc [C,H] bf16 (each C-column contiguous in k)
__global__ void cba_convert_V(const float* __restrict__ V, unsigned short* __restrict__ Vc) {
  __shared__ float tile[32][33];
  const int c0 = blockIdx.x * 32, h0 = blockIdx.y * 32;
  const int x = threadIdx.x, y = threadIdx.y;     // 32 x 8
#pragma unroll
  for (int i = 0; i < 4; ++i) {
    int r = y + 8 * i;
    tile[r][x] = V[(size_t)(h0 + r) * C_ + (c0 + x)];
  }
  __syncthreads();
#pragma unroll
  for (int i = 0; i < 4; ++i) {
    int r = y + 8 * i;                            // local c index
    Vc[(size_t)(c0 + r) * H_ + (h0 + x)] = f2bf(tile[x][r]);
  }
}

// dec_proj[b][c] = bias[c] + sum_h dec[b][h] * W[h][c]   (bias folded in)
__global__ void cba_dec_proj(const float* __restrict__ dec, const float* __restrict__ W,
                             const float* __restrict__ bias, float* __restrict__ dp) {
  const int b = blockIdx.x;
  const int c = threadIdx.x;                      // 1024 threads
  float acc = bias[c];
  const float* dv = dec + (size_t)b * H_;
  for (int h = 0; h < H_; ++h)
    acc = fmaf(dv[h], W[(size_t)h * C_ + c], acc);
  dp[b * C_ + c] = acc;
}

// Fused: scores[b][t] = sum_c tanh( (enc[b,t,:] @ V)[c] + dec_proj[b][c] ) * w[c]
union Frag { v16bf v; uint4 u[2]; };

__global__ void __launch_bounds__(256)
cba_scores(const float* __restrict__ enc, const unsigned short* __restrict__ Vc,
           const float* __restrict__ dp, const float* __restrict__ w,
           float* __restrict__ scores) {
  __shared__ unsigned short A_lds[16 * LDS_STRIDE];
  __shared__ float s_part[8][16];

  const int b   = blockIdx.x >> 7;                // / (T/16)
  const int t0  = (blockIdx.x & 127) << 4;        // * 16
  const int tid = threadIdx.x;

  for (int i = tid; i < 16 * H_; i += 256) {
    int r = i >> 10, h = i & (H_ - 1);
    A_lds[r * LDS_STRIDE + h] = f2bf(enc[((size_t)b * T_ + (t0 + r)) * H_ + h]);
  }
  __syncthreads();

  const int lane = tid & 31;
  const int wave = tid >> 5;
  const int nrow = lane & 15;                     // A row m / B col n / D col n
  const int half = lane >> 4;

  v8f zero;
#pragma unroll
  for (int i = 0; i < 8; ++i) zero[i] = 0.0f;
  v8f acc[8];
#pragma unroll
  for (int ct = 0; ct < 8; ++ct) acc[ct] = zero;

  for (int ks = 0; ks < 32; ++ks) {               // K = 1024 in steps of 32
    const int kb = ks * 32;
    Frag a;
    const unsigned short* ap = &A_lds[nrow * LDS_STRIDE + kb + half * 8];
    a.u[0] = *(const uint4*)(ap);                 // K = kb +      half*8 + 0..7
    a.u[1] = *(const uint4*)(ap + 16);            // K = kb + 16 + half*8 + 0..7
#pragma unroll
    for (int ct = 0; ct < 8; ++ct) {
      const unsigned short* bp =
          Vc + (size_t)((((wave << 3) + ct) << 4) + nrow) * H_ + kb + (half << 3);
      Frag bf;
      bf.u[0] = *(const uint4*)(bp);
      bf.u[1] = *(const uint4*)(bp + 16);
      acc[ct] = __builtin_amdgcn_wmma_f32_16x16x32_bf16(
          false, a.v, false, bf.v, (short)0, acc[ct], false, false);
    }
  }

  float ps[8];
#pragma unroll
  for (int v = 0; v < 8; ++v) ps[v] = 0.0f;
#pragma unroll
  for (int ct = 0; ct < 8; ++ct) {
    const int c = (((wave << 3) + ct) << 4) + nrow;
    const float d  = dp[b * C_ + c];
    const float wc = w[c];
#pragma unroll
    for (int v = 0; v < 8; ++v)
      ps[v] += tanhf(acc[ct][v] + d) * wc;        // row m = v + 8*half, col = c
  }
#pragma unroll
  for (int v = 0; v < 8; ++v) {
    float s = ps[v];
    s += __shfl_xor(s, 8, 16);
    s += __shfl_xor(s, 4, 16);
    s += __shfl_xor(s, 2, 16);
    s += __shfl_xor(s, 1, 16);
    if (nrow == 0) s_part[wave][v + (half << 3)] = s;
  }
  __syncthreads();
  if (tid < 16) {
    float s = 0.0f;
#pragma unroll
    for (int wv = 0; wv < 8; ++wv) s += s_part[wv][tid];  // fixed order: deterministic
    scores[b * T_ + t0 + tid] = s;
  }
}

__global__ void cba_softmax(const float* __restrict__ scores, float* __restrict__ align) {
  __shared__ float red[256];
  const int b = blockIdx.x, tid = threadIdx.x;
  const float* s = scores + b * T_;
  float mx = -3.4e38f;
  for (int t = tid; t < T_; t += 256) mx = fmaxf(mx, s[t]);
  red[tid] = mx; __syncthreads();
  for (int o = 128; o > 0; o >>= 1) {
    if (tid < o) red[tid] = fmaxf(red[tid], red[tid + o]);
    __syncthreads();
  }
  mx = red[0]; __syncthreads();
  float sum = 0.0f;
  for (int t = tid; t < T_; t += 256) sum += __expf(s[t] - mx);
  red[tid] = sum; __syncthreads();
  for (int o = 128; o > 0; o >>= 1) {
    if (tid < o) red[tid] += red[tid + o];
    __syncthreads();
  }
  const float inv = 1.0f / red[0];
  for (int t = tid; t < T_; t += 256) align[b * T_ + t] = __expf(s[t] - mx) * inv;
}

__global__ void cba_context_partial(const float* __restrict__ enc,
                                    const float* __restrict__ align,
                                    float* __restrict__ partial) {
  const int chunk = blockIdx.x, b = blockIdx.y, tid = threadIdx.x;  // 256 threads
  float a0 = 0.f, a1 = 0.f, a2 = 0.f, a3 = 0.f;
  const int t0 = chunk * 256;
  for (int tt = 0; tt < 256; ++tt) {
    const int t = t0 + tt;
    const float a = align[b * T_ + t];
    const float* row = enc + ((size_t)b * T_ + t) * H_;
    a0 = fmaf(a, row[tid      ], a0);
    a1 = fmaf(a, row[tid + 256], a1);
    a2 = fmaf(a, row[tid + 512], a2);
    a3 = fmaf(a, row[tid + 768], a3);
  }
  float* p = partial + (size_t)(b * 8 + chunk) * H_;
  p[tid] = a0; p[tid + 256] = a1; p[tid + 512] = a2; p[tid + 768] = a3;
}

__global__ void cba_context_reduce(const float* __restrict__ partial, float* __restrict__ out) {
  const int i = blockIdx.x * 256 + threadIdx.x;   // over B_*H_
  const int b = i >> 10, h = i & (H_ - 1);
  float s = 0.0f;
#pragma unroll
  for (int ch = 0; ch < 8; ++ch) s += partial[(size_t)(b * 8 + ch) * H_ + h];
  out[i] = s;
}

extern "C" void kernel_launch(void* const* d_in, const int* in_sizes, int n_in,
                              void* d_out, int out_size, void* d_ws, size_t ws_size,
                              hipStream_t stream) {
  const float* dec  = (const float*)d_in[0];
  const float* enc  = (const float*)d_in[1];
  const float* W    = (const float*)d_in[2];
  const float* V    = (const float*)d_in[3];
  const float* bias = (const float*)d_in[4];
  const float* w    = (const float*)d_in[5];
  float* out = (float*)d_out;

  char* ws = (char*)d_ws;
  unsigned short* Vc = (unsigned short*)(ws);                     // 2 MB
  float* dp      = (float*)(ws + (2u << 20));                     // 128 KB
  float* scores  = (float*)(ws + (2u << 20) + (128u << 10));      // 256 KB
  float* align   = (float*)(ws + (2u << 20) + (384u << 10));      // 256 KB
  float* partial = (float*)(ws + (2u << 20) + (640u << 10));      // 1 MB

  cba_convert_V<<<dim3(C_ / 32, H_ / 32), dim3(32, 8), 0, stream>>>(V, Vc);
  cba_dec_proj<<<dim3(B_), dim3(C_), 0, stream>>>(dec, W, bias, dp);
  cba_scores<<<dim3(B_ * (T_ / 16)), dim3(256), 0, stream>>>(enc, Vc, dp, w, scores);
  cba_softmax<<<dim3(B_), dim3(256), 0, stream>>>(scores, align);
  cba_context_partial<<<dim3(T_ / 256, B_), dim3(256), 0, stream>>>(enc, align, partial);
  cba_context_reduce<<<dim3((B_ * H_) / 256), dim3(256), 0, stream>>>(partial, out);
}
